// HierarchicalFusionAlpha_47502338294425
// MI455X (gfx1250) — matrix operand order repaired and checked by reading, not compiled, not run on hardware
//
#include <hip/hip_runtime.h>
#include <hip/hip_bf16.h>

#define NNODES   50000
#define IN_F     128
#define HID_F    128
#define OUT_F    16
#define NHEADS   4
#define NEL      800000
#define NEGE     200000
#define NEG_SLOPE 0.2f
#define MTILES   (NNODES / 16)   // 3125, exact

typedef __attribute__((ext_vector_type(16))) _Float16 v16h;
typedef __attribute__((ext_vector_type(8)))  _Float16 v8h;
typedef __attribute__((ext_vector_type(2)))  _Float16 v2h;
typedef __attribute__((ext_vector_type(8)))  float    v8f;
typedef __attribute__((ext_vector_type(4)))  float    v4f;
typedef __attribute__((ext_vector_type(4)))  unsigned int u32x4;
typedef __attribute__((ext_vector_type(8)))  int      i32x8;
typedef __attribute__((ext_vector_type(4)))  int      i32x4;

__device__ __forceinline__ float lrelu(float v) { return v > 0.f ? v : NEG_SLOPE * v; }

__device__ __forceinline__ void atomicMaxF(float* addr, float val) {
    unsigned int* ua = (unsigned int*)addr;
    unsigned int old = *ua;
    while (true) {
        float fold = __uint_as_float(old);
        if (fold >= val) break;
        unsigned int assumed = old;
        old = atomicCAS(ua, assumed, __float_as_uint(val));
        if (old == assumed) break;
    }
}

// ---------------------------------------------------------------------------
// Weight swizzle: row-major f32 W[K][Nout] -> f16 WMMA B fragments.
// B layout (16x16x32 f16, wave32): lane L holds column N; lane group g = L>>4;
// VGPR v (pair p) holds K = g*8 + 2v + p (+8 if v>=4).  Fragment storage:
// Bfrag[((nt*nKt + kt)*32 + lane)*16 + 2v + p]  -> one contiguous v16h per lane,
// and the whole (nt,kt) slab is contiguous => TDM-friendly 1-D copy.
// ---------------------------------------------------------------------------
__global__ void swizzle_weight(const float* __restrict__ W, int K, int Nout,
                               _Float16* __restrict__ Bfrag) {
    int gid = blockIdx.x * blockDim.x + threadIdx.x;
    int nKt = K >> 5;
    int nNt = Nout >> 4;
    int total = nNt * nKt * 32 * 8;
    if (gid >= total) return;
    int v    = gid & 7;
    int lane = (gid >> 3) & 31;
    int kt   = (gid >> 8) % nKt;
    int nt   = (gid >> 8) / nKt;
    int g    = lane >> 4;
    int n    = nt * 16 + (lane & 15);
#pragma unroll
    for (int p = 0; p < 2; ++p) {
        int k = g * 8 + 2 * v + p + (v >= 4 ? 8 : 0);
        Bfrag[(((size_t)nt * nKt + kt) * 32 + lane) * 16 + 2 * v + p] =
            (_Float16)W[(size_t)(kt * 32 + k) * Nout + n];
    }
}

// ---------------------------------------------------------------------------
// Wave-level WMMA GEMM with column-tile blocking and TDM-staged B operands.
//   C[M,Nout] = relu?(A[M,K] @ W + bias)
// - 4 waves/block, each wave owns one 16-row tile and NT 16-col tiles.
// - Wave 0 DMAs the block's B-fragment slab (NT*K/32 KB) into LDS with
//   tensor_load_to_lds (1-D descriptor, 8-byte units), s_wait_tensorcnt,
//   then a block barrier; all waves read B via ds_load_b128.
// - A fragment (16x32 f16): lane L, g=L>>4, M=L&15 holds two contiguous
//   8-half runs K=8g..8g+7 and K=16+8g..16+8g+7  -> two b128 loads.
// - D layout: VGPR r -> row = 16*tile + 8g + r, col = 16*ct + (L&15).
// ---------------------------------------------------------------------------
template <int K, int NT, bool A_IS_F32>
__global__ void gemm_wmma(const void* __restrict__ Aptr, int Mtiles,
                          const _Float16* __restrict__ Bfrag, int Nout,
                          const float* __restrict__ bias, int do_relu,
                          float* __restrict__ Cf,
                          _Float16* __restrict__ Ch, int chStride, int chOff) {
    constexpr int nKt = K >> 5;
    constexpr unsigned nUnits = (unsigned)NT * nKt * 128;  // 8-byte units in B slab
    __shared__ alignas(16) _Float16 shB[NT * nKt * 512];

    const int lane    = threadIdx.x & 31;
    const int wave    = threadIdx.x >> 5;
    const int rowTile = blockIdx.x * 4 + wave;
    const int colBase = blockIdx.y * NT;
    const int g = lane >> 4;
    const int m = lane & 15;

    // ---- stage B fragments into LDS with the Tensor Data Mover ----
    if (wave == 0) {
        unsigned long long ga =
            (unsigned long long)(uintptr_t)(Bfrag + (size_t)colBase * nKt * 512);
        unsigned lds_off = (unsigned)(uintptr_t)(void*)shB;  // LDS aperture low bits
        // D# group0: count=1 | lds_addr | global_addr[56:0] | type=2
        u32x4 g0 = { 1u, lds_off, (unsigned)(ga & 0xFFFFFFFFu),
                     (unsigned)((ga >> 32) & 0x1FFFFFFu) | 0x80000000u };
        // D# group1: data_size=8B; tensor_dim0=tile_dim0=nUnits; dim1=1; stride=nUnits
        i32x8 g1 = { (int)0x30000,
                     (int)((nUnits & 0xFFFFu) << 16),
                     (int)((nUnits >> 16) | (1u << 16)),
                     (int)(nUnits << 16),
                     1, (int)nUnits, 0, 0 };
        i32x4 g2 = { 0, 0, 0, 0 };
        i32x4 g3 = { 0, 0, 0, 0 };
        i32x8 g4 = { 0, 0, 0, 0, 0, 0, 0, 0 };
        __builtin_amdgcn_tensor_load_to_lds(g0, g1, g2, g3, g4, 0);
        __builtin_amdgcn_s_wait_tensorcnt((short)0);
    }
    __syncthreads();
    if (rowTile >= Mtiles) return;          // wave-uniform: EXEC stays all-1s

    const size_t rowBase = (size_t)(rowTile * 16 + m) * (size_t)K;

    v8f acc[NT];
#pragma unroll
    for (int t = 0; t < NT; ++t) acc[t] = (v8f){};

#pragma unroll
    for (int kt = 0; kt < nKt; ++kt) {
        v16h a;
        if constexpr (A_IS_F32) {
            const float* ar = (const float*)Aptr + rowBase + kt * 32 + 8 * g;
            v4f p0 = *(const v4f*)(ar);
            v4f p1 = *(const v4f*)(ar + 4);
            v4f p2 = *(const v4f*)(ar + 16);
            v4f p3 = *(const v4f*)(ar + 20);
#pragma unroll
            for (int i = 0; i < 4; ++i) {
                a[i]      = (_Float16)p0[i];
                a[4 + i]  = (_Float16)p1[i];
                a[8 + i]  = (_Float16)p2[i];
                a[12 + i] = (_Float16)p3[i];
            }
        } else {
            const _Float16* ar = (const _Float16*)Aptr + rowBase + kt * 32 + 8 * g;
            v8h lo = *(const v8h*)(ar);        // K = 8g .. 8g+7
            v8h hi = *(const v8h*)(ar + 16);   // K = 16+8g .. 16+8g+7
#pragma unroll
            for (int i = 0; i < 8; ++i) { a[i] = lo[i]; a[8 + i] = hi[i]; }
        }
#pragma unroll
        for (int t = 0; t < NT; ++t) {
            v16h b = *(const v16h*)(shB + ((size_t)(t * nKt + kt) * 32 + lane) * 16);
            acc[t] = __builtin_amdgcn_wmma_f32_16x16x32_f16(false, a, false, b,
                                                            (short)0, acc[t], false, false);
        }
    }

#pragma unroll
    for (int t = 0; t < NT; ++t) {
#pragma unroll
        for (int r = 0; r < 8; ++r) {
            int row  = rowTile * 16 + g * 8 + r;
            int col  = (colBase + t) * 16 + m;
            float vv = acc[t][r];
            if (bias) vv += bias[col];
            if (do_relu) vv = vv > 0.f ? vv : 0.f;
            if (Cf) Cf[(size_t)row * Nout + col] = vv;
            if (Ch) Ch[(size_t)row * chStride + chOff + col] = (_Float16)vv;
        }
    }
}

// --------------------------- GCN pieces ------------------------------------
__global__ void fill_f32(float* __restrict__ p, float val, int n) {
    int i = blockIdx.x * blockDim.x + threadIdx.x;
    if (i < n) p[i] = val;
}

__global__ void deg_edges(const int* __restrict__ dst, float* __restrict__ deg, int ne) {
    int i = blockIdx.x * blockDim.x + threadIdx.x;
    if (i < ne) atomicAdd(&deg[dst[i]], 1.0f);
}

__global__ void deg_rsqrt(const float* __restrict__ deg, float* __restrict__ dinv, int n) {
    int i = blockIdx.x * blockDim.x + threadIdx.x;
    if (i < n) dinv[i] = rsqrtf(fmaxf(deg[i], 1e-12f));
}

__global__ void gcn_self(const float* __restrict__ xwg, const float* __restrict__ dinv,
                         float* __restrict__ agg) {
    int i = blockIdx.x * blockDim.x + threadIdx.x;
    if (i >= NNODES * HID_F) return;
    float di = dinv[i >> 7];
    agg[i] = xwg[i] * di * di;
}

__global__ void gcn_edges(const int* __restrict__ src, const int* __restrict__ dst,
                          const float* __restrict__ xwg, const float* __restrict__ dinv,
                          float* __restrict__ agg) {
    int gid = blockIdx.x * blockDim.x + threadIdx.x;  // NEL*32 threads
    int e = gid >> 5;
    if (e >= NEL) return;
    int lane = gid & 31;
    int s = src[e], d = dst[e];
    float nrm = dinv[s] * dinv[d];
    int c = lane * 4;
    v4f xv = *(const v4f*)(xwg + (size_t)s * HID_F + c);
    float* ap = agg + (size_t)d * HID_F + c;
    atomicAdd(ap + 0, xv.x * nrm);
    atomicAdd(ap + 1, xv.y * nrm);
    atomicAdd(ap + 2, xv.z * nrm);
    atomicAdd(ap + 3, xv.w * nrm);
}

__global__ void gcn_fin(const float* __restrict__ agg, const float* __restrict__ bg,
                        _Float16* __restrict__ xcomb) {
    int i = blockIdx.x * blockDim.x + threadIdx.x;
    if (i >= NNODES * HID_F) return;
    int n = i >> 7, c = i & 127;
    float vv = agg[i] + bg[c];
    vv = vv > 0.f ? vv : 0.f;
    xcomb[(size_t)n * 256 + c] = (_Float16)vv;
}

// --------------------------- GAT pieces ------------------------------------
__global__ void gat_scores(const float* __restrict__ xwa, const float* __restrict__ asr,
                           const float* __restrict__ adst, float* __restrict__ a_s,
                           float* __restrict__ a_d) {
    int i = blockIdx.x * blockDim.x + threadIdx.x;  // N*4
    if (i >= NNODES * NHEADS) return;
    int n = i >> 2, h = i & 3;
    const float* xr = xwa + (size_t)n * 512 + h * 128;
    const float* av = asr + h * 128;
    const float* bv = adst + h * 128;
    float s0 = 0.f, s1 = 0.f;
    for (int c = 0; c < 128; c += 4) {
        v4f xv = *(const v4f*)(xr + c);
        v4f av4 = *(const v4f*)(av + c);
        v4f bv4 = *(const v4f*)(bv + c);
        s0 += xv.x * av4.x + xv.y * av4.y + xv.z * av4.z + xv.w * av4.w;
        s1 += xv.x * bv4.x + xv.y * bv4.y + xv.z * bv4.z + xv.w * bv4.w;
    }
    a_s[i] = s0;
    a_d[i] = s1;
}

__global__ void gat_emax_init(const float* __restrict__ a_s, const float* __restrict__ a_d,
                              float* __restrict__ emax) {
    int i = blockIdx.x * blockDim.x + threadIdx.x;
    if (i < NNODES * NHEADS) emax[i] = lrelu(a_s[i] + a_d[i]);   // self loop
}

__global__ void gat_emax_edges(const int* __restrict__ src, const int* __restrict__ dst,
                               const float* __restrict__ a_s, const float* __restrict__ a_d,
                               float* __restrict__ emax) {
    int gid = blockIdx.x * blockDim.x + threadIdx.x;  // NEGE*4
    int e = gid >> 2;
    if (e >= NEGE) return;
    int h = gid & 3;
    int s = src[e], d = dst[e];
    atomicMaxF(&emax[d * 4 + h], lrelu(a_s[s * 4 + h] + a_d[d * 4 + h]));
}

__global__ void gat_denom_init(const float* __restrict__ a_s, const float* __restrict__ a_d,
                               const float* __restrict__ emax, float* __restrict__ denom) {
    int i = blockIdx.x * blockDim.x + threadIdx.x;
    if (i < NNODES * NHEADS) denom[i] = expf(lrelu(a_s[i] + a_d[i]) - emax[i]);
}

__global__ void gat_denom_edges(const int* __restrict__ src, const int* __restrict__ dst,
                                const float* __restrict__ a_s, const float* __restrict__ a_d,
                                const float* __restrict__ emax, float* __restrict__ denom) {
    int gid = blockIdx.x * blockDim.x + threadIdx.x;  // NEGE*4
    int e = gid >> 2;
    if (e >= NEGE) return;
    int h = gid & 3;
    int s = src[e], d = dst[e];
    float ev = lrelu(a_s[s * 4 + h] + a_d[d * 4 + h]);
    atomicAdd(&denom[d * 4 + h], expf(ev - emax[d * 4 + h]));
}

__global__ void gat_agg_init(const float* __restrict__ xwa, const float* __restrict__ a_s,
                             const float* __restrict__ a_d, const float* __restrict__ emax,
                             const float* __restrict__ denom, float* __restrict__ outa) {
    int i = blockIdx.x * blockDim.x + threadIdx.x;  // N*512
    if (i >= NNODES * 512) return;
    int n = i >> 9, h = (i & 511) >> 7;
    int id4 = n * 4 + h;
    float al = expf(lrelu(a_s[id4] + a_d[id4]) - emax[id4]) / (denom[id4] + 1e-16f);
    outa[i] = xwa[i] * al;
}

__global__ void gat_agg_edges(const int* __restrict__ src, const int* __restrict__ dst,
                              const float* __restrict__ xwa, const float* __restrict__ a_s,
                              const float* __restrict__ a_d, const float* __restrict__ emax,
                              const float* __restrict__ denom, float* __restrict__ outa) {
    int gid = blockIdx.x * blockDim.x + threadIdx.x;  // NEGE*128
    int e = gid >> 7;
    if (e >= NEGE) return;
    int r = gid & 127, h = r >> 5, lane = r & 31;
    int s = src[e], d = dst[e];
    int id4 = d * 4 + h;
    float ev = lrelu(a_s[s * 4 + h] + a_d[d * 4 + h]);
    float w  = expf(ev - emax[id4]) / (denom[id4] + 1e-16f);
    int c = lane * 4;
    v4f xv = *(const v4f*)(xwa + (size_t)s * 512 + h * 128 + c);
    float* ap = outa + (size_t)d * 512 + h * 128 + c;
    atomicAdd(ap + 0, xv.x * w);
    atomicAdd(ap + 1, xv.y * w);
    atomicAdd(ap + 2, xv.z * w);
    atomicAdd(ap + 3, xv.w * w);
}

__global__ void gat_fin(const float* __restrict__ outa, const float* __restrict__ ba,
                        _Float16* __restrict__ xcomb) {
    int i = blockIdx.x * blockDim.x + threadIdx.x;
    if (i >= NNODES * HID_F) return;
    int n = i >> 7, c = i & 127;
    const float* p = outa + (size_t)n * 512 + c;
    float vv = (p[0] + p[128] + p[256] + p[384]) * 0.25f + ba[c];
    vv = vv > 0.f ? vv : 0.f;
    xcomb[(size_t)n * 256 + 128 + c] = (_Float16)vv;
}

// --------------------------- mean + head -----------------------------------
__global__ void colsum_reduce(const float* __restrict__ xp, float* __restrict__ cs) {
    int c  = threadIdx.x;           // 128
    int r0 = blockIdx.x * 256;
    float s = 0.f;
    for (int r = 0; r < 256; ++r) {
        int row = r0 + r;
        if (row < NNODES) s += xp[(size_t)row * 128 + c];
    }
    atomicAdd(&cs[c], s);
}

__global__ void head_kernel(const float* __restrict__ cs, const float* __restrict__ Wc1,
                            const float* __restrict__ bc1, const float* __restrict__ Wc2,
                            const float* __restrict__ bc2, const float* __restrict__ Wu,
                            const float* __restrict__ bu, float* __restrict__ out) {
    __shared__ float xf[128];
    __shared__ float h1[64];
    int t = threadIdx.x;  // 64 threads
    const float invN = 1.0f / (float)NNODES;
    xf[t]      = cs[t] * invN;
    xf[t + 64] = cs[t + 64] * invN;
    __syncthreads();
    float s = bc1[t];
    for (int c = 0; c < 128; ++c) s += xf[c] * Wc1[c * 64 + t];
    h1[t] = s > 0.f ? s : 0.f;
    __syncthreads();
    if (t < 16) {
        float p = bc2[t];
        for (int j = 0; j < 64; ++j) p += h1[j] * Wc2[j * 16 + t];
        out[t] = p;
        float u = bu[t];
        for (int c = 0; c < 128; ++c) u += xf[c] * Wu[c * 16 + t];
        out[16 + t] = 1.0f / (1.0f + expf(-u));
    }
}

// ---------------------------------------------------------------------------
extern "C" void kernel_launch(void* const* d_in, const int* in_sizes, int n_in,
                              void* d_out, int out_size, void* d_ws, size_t ws_size,
                              hipStream_t stream) {
    const float* x   = (const float*)d_in[0];
    const int*   lei = (const int*)d_in[1];
    const int*   gei = (const int*)d_in[2];
    const float* W1  = (const float*)d_in[3];
    const float* b1  = (const float*)d_in[4];
    const float* Wg  = (const float*)d_in[5];
    const float* bg  = (const float*)d_in[6];
    const float* Wa  = (const float*)d_in[7];
    const float* asr = (const float*)d_in[8];
    const float* ads = (const float*)d_in[9];
    const float* ba  = (const float*)d_in[10];
    const float* Wp  = (const float*)d_in[11];
    const float* bp  = (const float*)d_in[12];
    const float* Wc1 = (const float*)d_in[13];
    const float* bc1 = (const float*)d_in[14];
    const float* Wc2 = (const float*)d_in[15];
    const float* bc2 = (const float*)d_in[16];
    const float* Wu  = (const float*)d_in[17];
    const float* bu  = (const float*)d_in[18];

    const int* lsrc = lei;           const int* ldst = lei + NEL;
    const int* gsrc = gei;           const int* gdst = gei + NEGE;

    // ---- workspace carve-out (256B aligned) ----
    char* wsb = (char*)d_ws;
    size_t off = 0;
    auto carve = [&](size_t bytes) -> char* {
        char* p = wsb + off;
        off += (bytes + 255) & ~(size_t)255;
        return p;
    };
    _Float16* B1f  = (_Float16*)carve(128 * 128 * 2);
    _Float16* Bgf  = (_Float16*)carve(128 * 128 * 2);
    _Float16* Baf  = (_Float16*)carve(128 * 512 * 2);
    _Float16* Bpf  = (_Float16*)carve(256 * 128 * 2);
    _Float16* XLh  = (_Float16*)carve((size_t)NNODES * 128 * 2);
    float*    XWg  = (float*)carve((size_t)NNODES * 128 * 4);
    float*    XWa  = (float*)carve((size_t)NNODES * 512 * 4);
    float*    a_s  = (float*)carve((size_t)NNODES * 4 * 4);
    float*    a_d  = (float*)carve((size_t)NNODES * 4 * 4);
    float*    deg  = (float*)carve((size_t)NNODES * 4);
    float*    dinv = (float*)carve((size_t)NNODES * 4);
    float*    emax = (float*)carve((size_t)NNODES * 4 * 4);
    float*    deno = (float*)carve((size_t)NNODES * 4 * 4);
    float*    agg  = (float*)carve((size_t)NNODES * 128 * 4);
    float*    outa = (float*)carve((size_t)NNODES * 512 * 4);
    _Float16* XCh  = (_Float16*)carve((size_t)NNODES * 256 * 2);
    float*    csum = (float*)carve(128 * 4);
    float*    xpool = XWg;  // XWg is dead once GCN aggregation finishes

    const int B = 256;
    auto blks = [](int n, int b) { return (n + b - 1) / b; };

    // 1) weight swizzle to WMMA fragments (f16)
    swizzle_weight<<<blks(8 * 4 * 256, B), B, 0, stream>>>(W1, 128, 128, B1f);
    swizzle_weight<<<blks(8 * 4 * 256, B), B, 0, stream>>>(Wg, 128, 128, Bgf);
    swizzle_weight<<<blks(32 * 4 * 256, B), B, 0, stream>>>(Wa, 128, 512, Baf);
    swizzle_weight<<<blks(8 * 8 * 256, B), B, 0, stream>>>(Wp, 256, 128, Bpf);

    // 2) x_local = relu(x @ W1 + b1)  -> f16 (feeds next GEMMs only)
    dim3 gb(128);
    dim3 gridA((MTILES + 3) / 4, 4);   // Nout=128, NT=2 -> 4 column blocks
    gemm_wmma<128, 2, true><<<gridA, gb, 0, stream>>>(x, MTILES, B1f, 128, b1, 1,
                                                      nullptr, XLh, 128, 0);
    // 3) XWg = x_local @ Wg     4) XWa = x_local @ Wa
    gemm_wmma<128, 2, false><<<gridA, gb, 0, stream>>>(XLh, MTILES, Bgf, 128, nullptr, 0,
                                                       XWg, nullptr, 0, 0);
    dim3 gridW((MTILES + 3) / 4, 8);   // Nout=512, NT=4 -> 8 column blocks
    gemm_wmma<128, 4, false><<<gridW, gb, 0, stream>>>(XLh, MTILES, Baf, 512, nullptr, 0,
                                                       XWa, nullptr, 0, 0);

    // 5) GAT attention scores
    gat_scores<<<blks(NNODES * 4, B), B, 0, stream>>>(XWa, asr, ads, a_s, a_d);

    // 6) GCN degrees (self loop => init 1)
    fill_f32<<<blks(NNODES, B), B, 0, stream>>>(deg, 1.0f, NNODES);
    deg_edges<<<blks(NEL, B), B, 0, stream>>>(ldst, deg, NEL);
    deg_rsqrt<<<blks(NNODES, B), B, 0, stream>>>(deg, dinv, NNODES);

    // 7) GCN aggregation -> x_lg (f16 into XComb[:, :128])
    gcn_self<<<blks(NNODES * 128, B), B, 0, stream>>>(XWg, dinv, agg);
    gcn_edges<<<blks(NEL * 32, B), B, 0, stream>>>(lsrc, ldst, XWg, dinv, agg);
    gcn_fin<<<blks(NNODES * 128, B), B, 0, stream>>>(agg, bg, XCh);

    // 8) GAT edge softmax + aggregation -> x_gg (f16 into XComb[:, 128:])
    gat_emax_init<<<blks(NNODES * 4, B), B, 0, stream>>>(a_s, a_d, emax);
    gat_emax_edges<<<blks(NEGE * 4, B), B, 0, stream>>>(gsrc, gdst, a_s, a_d, emax);
    gat_denom_init<<<blks(NNODES * 4, B), B, 0, stream>>>(a_s, a_d, emax, deno);
    gat_denom_edges<<<blks(NEGE * 4, B), B, 0, stream>>>(gsrc, gdst, a_s, a_d, emax, deno);
    gat_agg_init<<<blks(NNODES * 512, B), B, 0, stream>>>(XWa, a_s, a_d, emax, deno, outa);
    gat_agg_edges<<<blks(NEGE * 128, B), B, 0, stream>>>(gsrc, gdst, XWa, a_s, a_d, emax,
                                                         deno, outa);
    gat_fin<<<blks(NNODES * 128, B), B, 0, stream>>>(outa, ba, XCh);

    // 9) x_pooled = relu(x_comb @ Wp + bp)   (K=256 WMMA GEMM)
    gemm_wmma<256, 2, false><<<gridA, gb, 0, stream>>>(XCh, MTILES, Bpf, 128, bp, 1,
                                                       xpool, nullptr, 0, 0);

    // 10) column mean + head
    fill_f32<<<1, 128, 0, stream>>>(csum, 0.0f, 128);
    colsum_reduce<<<blks(NNODES, 256), 128, 0, stream>>>(xpool, csum);
    head_kernel<<<1, 64, 0, stream>>>(csum, Wc1, bc1, Wc2, bc2, Wu, bu, (float*)d_out);
}
